// Spectral2xDown_67224828117116
// MI455X (gfx1250) — matrix operand order repaired and checked by reading, not compiled, not run on hardware
//
#include <hip/hip_runtime.h>
#include <hip/hip_bf16.h>

// ---------------------------------------------------------------------------
// Spectral2xDown for MI455X (gfx1250): bf16 WMMA NT-GEMMs + TDM B-tile DMA.
// ---------------------------------------------------------------------------

typedef __attribute__((ext_vector_type(16))) __bf16 v16bf;
typedef __attribute__((ext_vector_type(8)))  __bf16 v8bf;
typedef __attribute__((ext_vector_type(8)))  float  f32x8;
typedef __attribute__((ext_vector_type(4)))  unsigned int u32x4;
typedef __attribute__((ext_vector_type(8)))  int i32x8;
typedef __attribute__((ext_vector_type(4)))  int i32x4;

union BF16x16 { v16bf v; v8bf h[2]; };

#define NB  2
#define NCI 256
#define NCO 256
#define NHHI 256
#define NWHI 512
#define NHLO 128
#define NWLO 256
#define NLMAX 128
#define NMMAX 129
#define NMP1  144            // m padded to mult of 16 for rFFT GEMM N-dim
#define NMP4  160            // m padded to mult of 32 for irFFT GEMM K-dim
#define NK4   320            // 2*NMP4 (real || imag concatenated along K)
#define NHW   (NHLO*NWLO)    // 32768

// ---------------------------------------------------------------------------
// TDM: DMA a [valid_rows x 32] bf16 tile (row stride ldb elems) into LDS.
// 2D Tensor DMA Descriptor per CDNA5 ISA.
//   group0: count=1 | lds_addr | global_addr | type=2
//   group1: data_size=1(2B), tensor_dim0=32, tensor_dim1=valid_rows (OOB rows
//           read back as zero), tile_dim0=32, tile_dim1=64, stride0=ldb
// ---------------------------------------------------------------------------
__device__ __forceinline__ void tdm_load_b_tile(const __bf16* gsrc,
                                                unsigned lds_addr,
                                                int valid_rows, int ldb_elems)
{
    const unsigned long long ga = (unsigned long long)gsrc;
    u32x4 g0;
    g0[0] = 1u;                                        // count=1 (user D#)
    g0[1] = lds_addr;                                  // LDS byte address
    g0[2] = (unsigned)(ga & 0xFFFFFFFFu);              // global_addr[31:0]
    g0[3] = (unsigned)((ga >> 32) & 0x01FFFFFFu)       // global_addr[56:32]
          | (2u << 30);                                // type=2 ("image")
    i32x8 g1;
    g1[0] = (1 << 16);                                 // data_size=1 (2 bytes)
    g1[1] = (int)(32u << 16);                          // tensor_dim0[15:0]=32
    g1[2] = (int)((unsigned)valid_rows << 16);         // tensor_dim1[15:0]
    g1[3] = (int)(32u << 16);                          // tile_dim0=32
    g1[4] = 64;                                        // tile_dim1=64, tile_dim2=0
    g1[5] = ldb_elems;                                 // tensor_dim0_stride[31:0]
    g1[6] = 0;                                         // stride hi / dim1_stride
    g1[7] = 0;
    i32x4 gz = {0, 0, 0, 0};                           // groups 2/3 unused (2D)
#if __clang_major__ >= 23
    i32x8 gz8 = {0, 0, 0, 0, 0, 0, 0, 0};
    __builtin_amdgcn_tensor_load_to_lds(g0, g1, gz, gz, gz8, 0);
#else
    __builtin_amdgcn_tensor_load_to_lds(g0, g1, gz, gz, 0);
#endif
}

// ---------------------------------------------------------------------------
// Generic NT GEMM:  D[M,N] = A[M,K] * Bt[N,K]   (bf16 in, f32 accum via WMMA)
//   block = 256 threads = 8 waves; block tile 128(M) x 64(N); wave tile 16x64.
//   B tile (64x32 bf16) DMA'd into double-buffered LDS by the TDM; wave 0
//   issues the tensor loads and gates on TENSORcnt; barriers publish to all.
// MODE 0: bf16 out, idx = z*oBatch + obase + row*ors + col*ocs
// MODE 1: f32  out, same indexing, optional bias[col]
// MODE 2: bf16 out, y-transpose scatter: row=(b,c,k) -> [b][k*256+col][c]
// ---------------------------------------------------------------------------
template <int MODE>
__global__ __launch_bounds__(256) void gemm_nt(
    const __bf16* __restrict__ A,  long aBatch, int lda,
    const __bf16* __restrict__ Bt, long bBatch, int ldb,
    void* __restrict__ Out, long oBatch, long obase, long ors, long ocs,
    const float* __restrict__ bias,
    int M, int N, int K)
{
    __shared__ __align__(16) __bf16 ldsB[2][64 * 32];

    const int tid  = threadIdx.x;
    const int wave = tid >> 5;
    const int lane = tid & 31;
    const int half = lane >> 4;     // 0: lanes 0-15, 1: lanes 16-31
    const int l16  = lane & 15;
    const long z   = blockIdx.z;

    const __bf16* __restrict__ Ab = A  + z * aBatch;
    const __bf16* __restrict__ Bb = Bt + z * bBatch;

    const int m0 = blockIdx.x * 128 + wave * 16;
    const int n0 = blockIdx.y * 64;

    f32x8 acc0 = {}, acc1 = {}, acc2 = {}, acc3 = {};

    // A fragment row for this lane (16-bit A 16x32 layout)
    const int arow  = m0 + l16;
    const long abase = (long)arow * lda;

    // TDM source/targets
    int valid = N - n0; if (valid > 64) valid = 64;
    const __bf16* bsrc = Bb + (long)n0 * ldb;
    const unsigned lds0 = (unsigned)(unsigned long long)&ldsB[0][0];
    const unsigned lds1 = (unsigned)(unsigned long long)&ldsB[1][0];

    const int steps = K >> 5;                 // K is a multiple of 32

    if (wave == 0)
        tdm_load_b_tile(bsrc, lds0, valid, ldb);

    for (int i = 0; i < steps; ++i) {
        const int k0 = i << 5;
        if (wave == 0) {
            if (i + 1 < steps) {
                // prefetch next B tile into the other buffer, keep 1 in flight
                tdm_load_b_tile(bsrc + (k0 + 32), (i & 1) ? lds0 : lds1,
                                valid, ldb);
                __builtin_amdgcn_s_wait_tensorcnt(1);
            } else {
                __builtin_amdgcn_s_wait_tensorcnt(0);
            }
        }
        __syncthreads();                      // publish TDM tile to all waves

        // prefetch next A stripe (global_prefetch_b8)
        __builtin_prefetch(&Ab[abase + k0 + 128], 0, 1);

        BF16x16 afrag;
        const __bf16* ap = &Ab[abase + k0 + half * 8];
        afrag.h[0] = *(const v8bf*)(ap);
        afrag.h[1] = *(const v8bf*)(ap + 16);

        const __bf16* lcur = &ldsB[i & 1][0];
        BF16x16 bf0, bf1, bf2, bf3;
        {
            const __bf16* bp = &lcur[(0 * 16 + l16) * 32 + half * 8];
            bf0.h[0] = *(const v8bf*)(bp); bf0.h[1] = *(const v8bf*)(bp + 16);
        }
        {
            const __bf16* bp = &lcur[(1 * 16 + l16) * 32 + half * 8];
            bf1.h[0] = *(const v8bf*)(bp); bf1.h[1] = *(const v8bf*)(bp + 16);
        }
        {
            const __bf16* bp = &lcur[(2 * 16 + l16) * 32 + half * 8];
            bf2.h[0] = *(const v8bf*)(bp); bf2.h[1] = *(const v8bf*)(bp + 16);
        }
        {
            const __bf16* bp = &lcur[(3 * 16 + l16) * 32 + half * 8];
            bf3.h[0] = *(const v8bf*)(bp); bf3.h[1] = *(const v8bf*)(bp + 16);
        }

        acc0 = __builtin_amdgcn_wmma_f32_16x16x32_bf16(false, afrag.v, false, bf0.v,
                                                       (short)0, acc0, false, false);
        acc1 = __builtin_amdgcn_wmma_f32_16x16x32_bf16(false, afrag.v, false, bf1.v,
                                                       (short)0, acc1, false, false);
        acc2 = __builtin_amdgcn_wmma_f32_16x16x32_bf16(false, afrag.v, false, bf2.v,
                                                       (short)0, acc2, false, false);
        acc3 = __builtin_amdgcn_wmma_f32_16x16x32_bf16(false, afrag.v, false, bf3.v,
                                                       (short)0, acc3, false, false);

        __syncthreads();  // all waves done reading buf[i&1] before its reuse
    }

    // --------------------------- epilogue ----------------------------------
    f32x8 accs[4] = { acc0, acc1, acc2, acc3 };
    (void)M;
#pragma unroll
    for (int j = 0; j < 4; ++j) {
        const int col = n0 + j * 16 + l16;
        if (col >= N) continue;
#pragma unroll
        for (int i = 0; i < 8; ++i) {
            const long row = m0 + half * 8 + i;   // f32 C/D layout: VGPR i
            const float v = accs[j][i];
            if constexpr (MODE == 0) {
                const long idx = z * oBatch + obase + row * ors + (long)col * ocs;
                ((__bf16*)Out)[idx] = (__bf16)v;
            } else if constexpr (MODE == 1) {
                const long idx = z * oBatch + obase + row * ors + (long)col * ocs;
                float o = v;
                if (bias) o += bias[col];
                ((float*)Out)[idx] = o;
            } else { // MODE 2: y-transpose scatter  row=(b,c,k) -> [b][k*W+col][c]
                const int b = (int)(row >> 15);
                const int c = (int)((row >> 7) & 255);
                const int k = (int)(row & 127);
                const long idx = (long)b * ((long)NHW * NCI)
                               + ((long)(k * NWLO + col)) * NCI + c;
                ((__bf16*)Out)[idx] = (__bf16)v;
            }
        }
    }
}

// ---------------------------------------------------------------------------
// Small helper kernels
// ---------------------------------------------------------------------------
__global__ __launch_bounds__(256) void k_cvt4(const float4* __restrict__ in,
                                              __bf16* __restrict__ out, long n4)
{
    const long i = (long)blockIdx.x * blockDim.x + threadIdx.x;
    if (i < n4) {
        const float4 v = in[i];
        const long o = i * 4;
        out[o + 0] = (__bf16)v.x; out[o + 1] = (__bf16)v.y;
        out[o + 2] = (__bf16)v.z; out[o + 3] = (__bf16)v.w;
    }
}

// rFFT basis (transposed): Br[m][w] = (2pi/512) cos(2pi m w/512), Bi = -(..) sin
__global__ __launch_bounds__(256) void k_basis1(__bf16* __restrict__ br,
                                                __bf16* __restrict__ bi)
{
    const int m = blockIdx.x;                 // 0..143 (>=129 zero-padded)
    const float scale = 6.283185307179586f / 512.0f;
    for (int w = threadIdx.x; w < NWHI; w += blockDim.x) {
        float vr = 0.f, vi = 0.f;
        if (m < NMMAX) {
            const int   q  = (m * w) & 511;
            const float th = 6.283185307179586f * (float)q / 512.0f;
            vr =  scale * cosf(th);
            vi = -scale * sinf(th);
        }
        br[m * NWHI + w] = (__bf16)vr;
        bi[m * NWHI + w] = (__bf16)vi;
    }
}

// irFFT basis (transposed, K = [real 160 | imag 160]):
//   y[w] = sum_m c_m ( gr[m] cos(2pi m w/256) - gi[m] sin(2pi m w/256) )
__global__ void k_basis4(__bf16* __restrict__ b4)
{
    const int w  = blockIdx.x;                // 0..255
    const int mm = threadIdx.x;               // 0..319
    if (mm >= NK4) return;
    float v = 0.f;
    const int isImag = (mm >= NMP4);
    const int m = isImag ? (mm - NMP4) : mm;
    if (m < NMMAX) {
        const float cm = (m == 0 || m == NMMAX - 1) ? 1.0f : 2.0f;
        const int   q  = (m * w) & 255;
        const float th = 6.283185307179586f * (float)q / 256.0f;
        v = isImag ? (-cm * sinf(th)) : (cm * cosf(th));
    }
    b4[(long)w * NK4 + mm] = (__bf16)v;
}

// pct_lo [m][l][k] -> transposed bf16 [m][k][l]
__global__ __launch_bounds__(256) void k_pct_t(const float* __restrict__ in,
                                               __bf16* __restrict__ out)
{
    const long n = (long)NMMAX * NLMAX * NHLO;
    const long i = (long)blockIdx.x * 256 + threadIdx.x;
    if (i < n) {
        const long m   = i >> 14;             // / (128*128)
        const int  rem = (int)(i & 16383);
        const int  l   = rem >> 7;
        const int  k   = rem & 127;
        out[m * 16384 + k * 128 + l] = (__bf16)in[i];
    }
}

// GroupNorm partial reduction: grid(64 bg, 128 chunks), deterministic (no atomics)
__global__ __launch_bounds__(256) void k_gn_partial(const float* __restrict__ h,
                                                    float* __restrict__ partials)
{
    const int bg = blockIdx.x;                // b*32+g
    const int ch = blockIdx.y;                // chunk 0..127
    const int b  = bg >> 5, g = bg & 31;
    const long base = (long)b * ((long)NHW * NCO) + g * 8;
    float s = 0.f, s2 = 0.f;
    const int e0 = ch * 2048 + threadIdx.x;
#pragma unroll
    for (int it = 0; it < 8; ++it) {
        const int e  = e0 + it * 256;         // 0..262143 within group
        const int hw = e >> 3, oo = e & 7;
        const float v = h[base + (long)hw * NCO + oo];
        s += v; s2 += v * v;
    }
    __shared__ float rs[256], rq[256];
    rs[threadIdx.x] = s; rq[threadIdx.x] = s2;
    __syncthreads();
    for (int off = 128; off > 0; off >>= 1) {
        if (threadIdx.x < off) {
            rs[threadIdx.x] += rs[threadIdx.x + off];
            rq[threadIdx.x] += rq[threadIdx.x + off];
        }
        __syncthreads();
    }
    if (threadIdx.x == 0) {
        partials[bg * 128 + ch]        = rs[0];
        partials[8192 + bg * 128 + ch] = rq[0];
    }
}

__global__ void k_gn_final(const float* __restrict__ partials,
                           float* __restrict__ stats)
{
    const int bg = threadIdx.x;
    if (bg < 64) {
        float s = 0.f, s2 = 0.f;
        for (int c = 0; c < 128; ++c) {
            s  += partials[bg * 128 + c];
            s2 += partials[8192 + bg * 128 + c];
        }
        const float inv = 1.0f / 262144.0f;
        const float mu  = s * inv;
        const float var = s2 * inv - mu * mu;
        stats[bg * 2 + 0] = mu;
        stats[bg * 2 + 1] = rsqrtf(var + 1e-5f);
    }
}

// normalize + affine + exact GELU + transpose store to [b][o][h][w]
__global__ __launch_bounds__(256) void k_gn_gelu(const float* __restrict__ h,
                                                 const float* __restrict__ stats,
                                                 const float* __restrict__ conv_b,
                                                 const float* __restrict__ gamma,
                                                 const float* __restrict__ beta,
                                                 float* __restrict__ out)
{
    const long i  = (long)blockIdx.x * 256 + threadIdx.x;   // 16,777,216
    const int  w  = (int)(i & 255);
    const int  kk = (int)((i >> 8) & 127);
    const int  o  = (int)((i >> 15) & 255);
    const int  b  = (int)(i >> 23);
    const int  g  = o >> 3;
    const long hidx = (long)b * ((long)NHW * NCO)
                    + ((long)(kk * NWLO + w)) * NCO + o;
    float v = h[hidx] + conv_b[o];
    const float mu = stats[(b * 32 + g) * 2 + 0];
    const float rs = stats[(b * 32 + g) * 2 + 1];
    v = (v - mu) * rs * gamma[o] + beta[o];
    out[i] = 0.5f * v * (1.0f + erff(v * 0.7071067811865475f));
}

// ---------------------------------------------------------------------------
// Launch
// ---------------------------------------------------------------------------
extern "C" void kernel_launch(void* const* d_in, const int* in_sizes, int n_in,
                              void* d_out, int out_size, void* d_ws, size_t ws_size,
                              hipStream_t stream)
{
    (void)in_sizes; (void)n_in; (void)out_size; (void)ws_size;
    const float* x       = (const float*)d_in[0]; // [2,256,256,512]
    const float* conv_w  = (const float*)d_in[1]; // [256,256]
    const float* conv_b  = (const float*)d_in[2]; // [256]
    const float* gamma   = (const float*)d_in[3]; // [256]
    const float* beta    = (const float*)d_in[4]; // [256]
    const float* wmat_hi = (const float*)d_in[5]; // [129,128,256]
    const float* pct_lo  = (const float*)d_in[6]; // [129,128,128]
    float* out = (float*)d_out;

    char* wsb = (char*)d_ws;
    size_t off = 0;
    auto take = [&](size_t bytes) -> char* {
        off = (off + 255) & ~(size_t)255;
        char* r = wsb + off;
        off += bytes;
        return r;
    };

    __bf16* xbf   = (__bf16*)take(134217728ull);           // [131072][512] bf16
    __bf16* b1r   = (__bf16*)take((size_t)NMP1 * NWHI * 2);
    __bf16* b1i   = (__bf16*)take((size_t)NMP1 * NWHI * 2);
    __bf16* Xr    = (__bf16*)take((size_t)NMP1 * 512 * NHHI * 2); // [m][bc][k]
    __bf16* Xi    = (__bf16*)take((size_t)NMP1 * 512 * NHHI * 2);
    __bf16* wmatb = (__bf16*)take((size_t)NMMAX * NLMAX * NHHI * 2);
    __bf16* pctt  = (__bf16*)take((size_t)NMMAX * NHLO * NLMAX * 2);
    __bf16* flmr  = (__bf16*)take((size_t)NMMAX * 512 * NLMAX * 2);
    __bf16* flmi  = (__bf16*)take((size_t)NMMAX * 512 * NLMAX * 2);
    __bf16* b4    = (__bf16*)take((size_t)NWLO * NK4 * 2);
    __bf16* cwb   = (__bf16*)take((size_t)NCO * NCI * 2);
    float*  parts = (float*)take(64 * 128 * 2 * 4);
    float*  stats = (float*)take(64 * 2 * 4);

    // aliases over dead regions:
    __bf16* gbuf = xbf;                                      // [65536][320], x dead after stage1
    __bf16* ytb  = (__bf16*)((char*)xbf + 50331648ull);      // [b][hw][c], 48MB offset
    float*  h_t  = (float*)Xr;                               // [b][hw][o], X dead after stage2

    // --- precompute / convert ---------------------------------------------
    k_cvt4<<<65536, 256, 0, stream>>>((const float4*)x, xbf, 16777216L);
    k_basis1<<<NMP1, 256, 0, stream>>>(b1r, b1i);
    k_cvt4<<<4128, 256, 0, stream>>>((const float4*)wmat_hi, wmatb, 1056768L);
    k_pct_t<<<8256, 256, 0, stream>>>(pct_lo, pctt);
    k_cvt4<<<64, 256, 0, stream>>>((const float4*)conv_w, cwb, 16384L);
    k_basis4<<<NWLO, NK4, 0, stream>>>(b4);

    // --- stage 1: rFFT as GEMM, scatter to per-m planes -------------------
    // M=131072, N=144, K=512; out idx = col*131072 + row (column-major planes)
    gemm_nt<0><<<dim3(1024, 3, 1), 256, 0, stream>>>(
        xbf, 0, NWHI, b1r, 0, NWHI, Xr, 0, 0, 1, 131072, nullptr, 131072, NMP1, NWHI);
    gemm_nt<0><<<dim3(1024, 3, 1), 256, 0, stream>>>(
        xbf, 0, NWHI, b1i, 0, NWHI, Xi, 0, 0, 1, 131072, nullptr, 131072, NMP1, NWHI);

    // --- stage 2: forward Legendre, batched over m=129 --------------------
    // per m: [512 x 256] * wmat[m]^T(128x256) -> flm[m][512][128]
    gemm_nt<0><<<dim3(4, 2, NMMAX), 256, 0, stream>>>(
        Xr, 131072, NHHI, wmatb, (long)NLMAX * NHHI, NHHI,
        flmr, (long)512 * NLMAX, 0, NLMAX, 1, nullptr, 512, NLMAX, NHHI);
    gemm_nt<0><<<dim3(4, 2, NMMAX), 256, 0, stream>>>(
        Xi, 131072, NHHI, wmatb, (long)NLMAX * NHHI, NHHI,
        flmi, (long)512 * NLMAX, 0, NLMAX, 1, nullptr, 512, NLMAX, NHHI);

    // --- stage 3: inverse Legendre, scatter into g[(bc*128+k)][320] -------
    // real -> cols [m], imag -> cols [160+m]
    gemm_nt<0><<<dim3(4, 2, NMMAX), 256, 0, stream>>>(
        flmr, (long)512 * NLMAX, NLMAX, pctt, (long)NHLO * NLMAX, NLMAX,
        gbuf, 1, 0, (long)NHLO * NK4, NK4, nullptr, 512, NHLO, NLMAX);
    gemm_nt<0><<<dim3(4, 2, NMMAX), 256, 0, stream>>>(
        flmi, (long)512 * NLMAX, NLMAX, pctt, (long)NHLO * NLMAX, NLMAX,
        gbuf, 1, NMP4, (long)NHLO * NK4, NK4, nullptr, 512, NHLO, NLMAX);

    // --- stage 4: irFFT as GEMM (K=320 = real||imag), y stored [b][hw][c] --
    gemm_nt<2><<<dim3(512, 4, 1), 256, 0, stream>>>(
        gbuf, 0, NK4, b4, 0, NK4, ytb, 0, 0, 0, 0, nullptr, 65536, NWLO, NK4);

    // --- stage 5: 1x1 conv per batch: h_t[b][hw][o] = y_t[b]·conv_w^T + b --
    gemm_nt<1><<<dim3(256, 4, NB), 256, 0, stream>>>(
        ytb, (long)NHW * NCI, NCI, cwb, 0, NCI,
        h_t, (long)NHW * NCO, 0, NCO, 1, conv_b, NHW, NCO, NCI);

    // --- stage 6: GroupNorm + GELU ----------------------------------------
    k_gn_partial<<<dim3(64, 128), 256, 0, stream>>>(h_t, parts);
    k_gn_final<<<1, 64, 0, stream>>>(parts, stats);
    k_gn_gelu<<<65536, 256, 0, stream>>>(h_t, stats, conv_b, gamma, beta, out);
}
